// SSC_GT_6382321402443
// MI455X (gfx1250) — matrix operand order repaired and checked by reading, not compiled, not run on hardware
//
#include <hip/hip_runtime.h>
#include <hip/hip_bf16.h>
#include <math.h>

typedef __attribute__((ext_vector_type(16))) _Float16 v16h;
typedef __attribute__((ext_vector_type(8)))  float    v8f;

#define DEVINL static __device__ __forceinline__

namespace {

constexpr int kB = 16, kN = 4096, kD = 512, kDN = 64, kM = 64;
constexpr int kLDB = 72;  // f16 LDS staging row stride (pad vs 64 to dodge bank conflicts)

DEVINL v8f wmma16(v16h a, v16h b, v8f c) {
  // D = A(16x32 f16) x B(32x16 f16) + C(16x16 f32)
  return __builtin_amdgcn_wmma_f32_16x16x32_f16(false, a, false, b, (short)0, c,
                                                false, false);
}

DEVINL v8f vzero8() {
  v8f z;
#pragma unroll
  for (int i = 0; i < 8; ++i) z[i] = 0.0f;
  return z;
}

// A-fragment: 16x32, lane half h holds contiguous K block 16h..16h+15.
// p must already point at rowptr + k0 + 16*half.
template <typename T>
DEVINL v16h load_afrag(const T* p) {
  v16h f;
#pragma unroll
  for (int i = 0; i < 8; ++i) f[i] = (_Float16)p[i];
#pragma unroll
  for (int i = 0; i < 8; ++i) f[8 + i] = (_Float16)p[8 + i];
  return f;
}

// B-fragment: 32x16 from W stored (N,K) row-major (computing X @ W^T).
// Lane half h holds contiguous K block 16h..16h+15 of row n = lane&15.
// p must already point at W + (n0 + r)*ld + k0 + 16*half.
template <typename T>
DEVINL v16h load_bfrag(const T* p) {
  v16h f;
#pragma unroll
  for (int i = 0; i < 16; ++i) f[i] = (_Float16)p[i];
  return f;
}

// ---------------------------------------------------------------- weights->f16
__global__ __launch_bounds__(256) void k_convert(
    const float* __restrict__ ws1, const float* __restrict__ wt2n,
    const float* __restrict__ wn2t, _Float16* __restrict__ ws1h,
    _Float16* __restrict__ wt2nh, _Float16* __restrict__ wn2th) {
  int i = blockIdx.x * 256 + threadIdx.x;  // 128*256 == 32768 == 64*512
  ws1h[i] = (_Float16)ws1[i];
  wt2nh[i] = (_Float16)wt2n[i];
  wn2th[i] = (_Float16)wn2t[i];
}

// ---------------------------------------------------------------- score MLP
// scores[bn] = w_s2 . relu(W_s1 x + b_s1) + b_s2 ; one wave per 16-token tile.
__global__ __launch_bounds__(256) void k_scores(
    const float* __restrict__ tok, const _Float16* __restrict__ w1h,
    const float* __restrict__ b1, const float* __restrict__ w2,
    const float* __restrict__ b2, float* __restrict__ scores) {
  int wave = threadIdx.x >> 5, lane = threadIdx.x & 31;
  int half = lane >> 4, r = lane & 15;
  int tile = blockIdx.x * 8 + wave;
  size_t g0 = (size_t)tile * 16;
  const float* arow = tok + (g0 + (size_t)r) * kD + 16 * half;

  v8f acc[4];
#pragma unroll
  for (int nt = 0; nt < 4; ++nt) acc[nt] = vzero8();

  for (int k0 = 0; k0 < kD; k0 += 32) {
    v16h a = load_afrag(arow + k0);
#pragma unroll
    for (int nt = 0; nt < 4; ++nt) {
      v16h bf = load_bfrag(w1h + (size_t)(nt * 16 + r) * kD + k0 + 16 * half);
      acc[nt] = wmma16(a, bf, acc[nt]);
    }
  }

  // relu + dot with w_s2 along the 64 hidden units, reduce across 16 lanes.
  float ps[8];
#pragma unroll
  for (int v = 0; v < 8; ++v) ps[v] = 0.0f;
#pragma unroll
  for (int nt = 0; nt < 4; ++nt) {
    float w2v = w2[nt * 16 + r];
    float b1v = b1[nt * 16 + r];
#pragma unroll
    for (int v = 0; v < 8; ++v) {
      float h = fmaxf(acc[nt][v] + b1v, 0.0f);
      ps[v] += h * w2v;
    }
  }
#pragma unroll
  for (int msk = 1; msk < 16; msk <<= 1) {
#pragma unroll
    for (int v = 0; v < 8; ++v) ps[v] += __shfl_xor(ps[v], msk, 16);
  }
  if (r == 0) {
    float bb = b2[0];
#pragma unroll
    for (int v = 0; v < 8; ++v) scores[g0 + 8 * half + v] = ps[v] + bb;
  }
}

// ---------------------------------------------------------------- top-64
__global__ __launch_bounds__(256) void k_topk(const float* __restrict__ scores,
                                              int* __restrict__ idx) {
  __shared__ float s[kN];
  __shared__ float rv[256];
  __shared__ int ri[256];
  int b = blockIdx.x, t = threadIdx.x;
  for (int i = t; i < kN; i += 256) s[i] = scores[(size_t)b * kN + i];
  __syncthreads();
  for (int it = 0; it < kM; ++it) {
    float bv = -3.0e38f;
    int bi = kN;
    for (int i = t; i < kN; i += 256) {
      float v = s[i];
      if (v > bv || (v == bv && i < bi)) { bv = v; bi = i; }
    }
    rv[t] = bv; ri[t] = bi;
    __syncthreads();
    for (int off = 128; off > 0; off >>= 1) {
      if (t < off) {
        float ov = rv[t + off]; int oi = ri[t + off];
        if (ov > rv[t] || (ov == rv[t] && oi < ri[t])) { rv[t] = ov; ri[t] = oi; }
      }
      __syncthreads();
    }
    if (t == 0) { idx[b * kM + it] = ri[0]; s[ri[0]] = -3.0e38f; }
    __syncthreads();
  }
}

// ---------------------------------------------------------------- graph (per batch)
// H0 = sel @ w_t2n^T (WMMA) ; cosine adjacency ; 2-layer GCN (f32 VALU in LDS).
__global__ __launch_bounds__(256) void k_graph(
    const float* __restrict__ tok, const int* __restrict__ idx,
    const _Float16* __restrict__ wt2nh, const float* __restrict__ wg1,
    const float* __restrict__ wg2, _Float16* __restrict__ Hg16,
    _Float16* __restrict__ HgT16) {
  __shared__ float sH[kM * kDN];
  __shared__ float sA[kM * kM];
  __shared__ float sT[kM * kDN];
  __shared__ float sInv[kM];
  __shared__ float sRs[kM];
  int b = blockIdx.x, t = threadIdx.x;
  int wave = t >> 5, lane = t & 31, half = lane >> 4, r = lane & 15;

  // ---- H0 via WMMA: wave w owns row-tile mt = w/2, col-tiles nt0, nt0+1 ----
  int mt = wave >> 1;
  int nt0 = (wave & 1) * 2;
  int trow = idx[b * kM + mt * 16 + r];
  const float* rowp = tok + ((size_t)b * kN + (size_t)trow) * kD + 16 * half;
  v8f a0 = vzero8(), a1 = vzero8();
  for (int k0 = 0; k0 < kD; k0 += 32) {
    v16h a = load_afrag(rowp + k0);
    v16h bf0 = load_bfrag(wt2nh + (size_t)(nt0 * 16 + r) * kD + k0 + 16 * half);
    v16h bf1 = load_bfrag(wt2nh + (size_t)((nt0 + 1) * 16 + r) * kD + k0 + 16 * half);
    a0 = wmma16(a, bf0, a0);
    a1 = wmma16(a, bf1, a1);
  }
#pragma unroll
  for (int v = 0; v < 8; ++v) {
    int row = mt * 16 + v + 8 * half;
    sH[row * kDN + nt0 * 16 + r] = a0[v];
    sH[row * kDN + (nt0 + 1) * 16 + r] = a1[v];
  }
  __syncthreads();

  // ---- inverse row norms ----
  if (t < kM) {
    float ss = 0.f;
    for (int j = 0; j < kDN; ++j) { float x = sH[t * kDN + j]; ss += x * x; }
    sInv[t] = 1.0f / fmaxf(sqrtf(ss), 1e-6f);
  }
  __syncthreads();

  // ---- A = relu(cos-sim) + I ----
  for (int e = t; e < kM * kM; e += 256) {
    int m = e >> 6, k = e & 63;
    float d = 0.f;
    for (int j = 0; j < kDN; ++j) d += sH[m * kDN + j] * sH[k * kDN + j];
    sA[e] = fmaxf(d * sInv[m] * sInv[k], 0.0f) + (m == k ? 1.0f : 0.0f);
  }
  __syncthreads();
  if (t < kM) {
    float s = 0.f;
    for (int j = 0; j < kM; ++j) s += sA[t * kM + j];
    sRs[t] = 1.0f / fmaxf(s, 1e-6f);
  }
  __syncthreads();
  for (int e = t; e < kM * kM; e += 256) sA[e] *= sRs[e >> 6];
  __syncthreads();

  // ---- T = A @ H0 ----
  for (int e = t; e < kM * kDN; e += 256) {
    int m = e >> 6, c = e & 63;
    float s = 0.f;
    for (int j = 0; j < kM; ++j) s += sA[m * kM + j] * sH[j * kDN + c];
    sT[e] = s;
  }
  __syncthreads();
  // ---- H = relu(T @ wg1^T) ----
  for (int e = t; e < kM * kDN; e += 256) {
    int m = e >> 6, c = e & 63;
    float s = 0.f;
    for (int j = 0; j < kDN; ++j) s += sT[m * kDN + j] * wg1[c * kDN + j];
    sH[e] = fmaxf(s, 0.0f);
  }
  __syncthreads();
  // ---- T = A @ H ----
  for (int e = t; e < kM * kDN; e += 256) {
    int m = e >> 6, c = e & 63;
    float s = 0.f;
    for (int j = 0; j < kM; ++j) s += sA[m * kM + j] * sH[j * kDN + c];
    sT[e] = s;
  }
  __syncthreads();
  // ---- Hg = relu(T @ wg2^T) -> global f16 (row-major and transposed) ----
  for (int e = t; e < kM * kDN; e += 256) {
    int m = e >> 6, c = e & 63;
    float s = 0.f;
    for (int j = 0; j < kDN; ++j) s += sT[m * kDN + j] * wg2[c * kDN + j];
    _Float16 hh = (_Float16)fmaxf(s, 0.0f);
    Hg16[(size_t)b * kM * kDN + m * kDN + c] = hh;
    HgT16[(size_t)b * kM * kDN + c * kM + m] = hh;
  }
}

// ---------------------------------------------------------------- inject + out
// Fused: proj -> logits -> softmax -> attn@Hg -> back-proj -> residual.
__global__ __launch_bounds__(256) void k_inject(
    const float* __restrict__ tok, const _Float16* __restrict__ wt2nh,
    const _Float16* __restrict__ wn2th, const _Float16* __restrict__ Hg16,
    const _Float16* __restrict__ HgT16, float* __restrict__ out) {
  __shared__ _Float16 sHg[kM * kDN];
  __shared__ _Float16 sHgT[kDN * kM];
  __shared__ _Float16 sBuf[8][16 * kLDB];
  int b = blockIdx.x >> 5, chunk = blockIdx.x & 31;
  int wave = threadIdx.x >> 5, lane = threadIdx.x & 31;
  int half = lane >> 4, r = lane & 15;

  for (int i = threadIdx.x; i < kM * kDN; i += 256) {
    sHg[i] = Hg16[(size_t)b * kM * kDN + i];
    sHgT[i] = HgT16[(size_t)b * kM * kDN + i];
  }
  __syncthreads();

  _Float16* buf = &sBuf[wave][0];
  size_t g0 = (size_t)b * kN + (size_t)chunk * 128 + (size_t)wave * 16;
  const float* tokb = tok + g0 * kD;
  const float* arow = tokb + (size_t)r * kD + 16 * half;

  // ---- proj = tok @ w_t2n^T  (16 x 64) ----
  v8f acc[4];
#pragma unroll
  for (int nt = 0; nt < 4; ++nt) acc[nt] = vzero8();
  for (int k0 = 0; k0 < kD; k0 += 32) {
    v16h a = load_afrag(arow + k0);
#pragma unroll
    for (int nt = 0; nt < 4; ++nt) {
      v16h bf = load_bfrag(wt2nh + (size_t)(nt * 16 + r) * kD + k0 + 16 * half);
      acc[nt] = wmma16(a, bf, acc[nt]);
    }
  }
#pragma unroll
  for (int nt = 0; nt < 4; ++nt)
#pragma unroll
    for (int v = 0; v < 8; ++v)
      buf[(v + 8 * half) * kLDB + nt * 16 + r] = (_Float16)acc[nt][v];
  __builtin_amdgcn_wave_barrier();

  // ---- logits = proj @ Hg^T / 8  (16 x 64) ----
  v8f lg[4];
#pragma unroll
  for (int nt = 0; nt < 4; ++nt) lg[nt] = vzero8();
#pragma unroll
  for (int kt = 0; kt < 2; ++kt) {
    v16h a = load_afrag(buf + r * kLDB + kt * 32 + 16 * half);
#pragma unroll
    for (int nt = 0; nt < 4; ++nt) {
      v16h bf = load_bfrag(sHg + (nt * 16 + r) * kDN + kt * 32 + 16 * half);
      lg[nt] = wmma16(a, bf, lg[nt]);
    }
  }
#pragma unroll
  for (int nt = 0; nt < 4; ++nt)
#pragma unroll
    for (int v = 0; v < 8; ++v) lg[nt][v] *= 0.125f;  // 1/sqrt(DN)

  // ---- softmax over 64 nodes (rows live across 16 lanes x 4 tiles) ----
  float rm[8], sm[8];
#pragma unroll
  for (int v = 0; v < 8; ++v)
    rm[v] = fmaxf(fmaxf(lg[0][v], lg[1][v]), fmaxf(lg[2][v], lg[3][v]));
#pragma unroll
  for (int msk = 1; msk < 16; msk <<= 1) {
#pragma unroll
    for (int v = 0; v < 8; ++v) rm[v] = fmaxf(rm[v], __shfl_xor(rm[v], msk, 16));
  }
#pragma unroll
  for (int v = 0; v < 8; ++v) sm[v] = 0.0f;
#pragma unroll
  for (int nt = 0; nt < 4; ++nt)
#pragma unroll
    for (int v = 0; v < 8; ++v) {
      float e = __expf(lg[nt][v] - rm[v]);
      lg[nt][v] = e;
      sm[v] += e;
    }
#pragma unroll
  for (int msk = 1; msk < 16; msk <<= 1) {
#pragma unroll
    for (int v = 0; v < 8; ++v) sm[v] += __shfl_xor(sm[v], msk, 16);
  }
#pragma unroll
  for (int v = 0; v < 8; ++v) sm[v] = 1.0f / sm[v];
#pragma unroll
  for (int nt = 0; nt < 4; ++nt)
#pragma unroll
    for (int v = 0; v < 8; ++v)
      buf[(v + 8 * half) * kLDB + nt * 16 + r] = (_Float16)(lg[nt][v] * sm[v]);
  __builtin_amdgcn_wave_barrier();

  // ---- injected = attn @ Hg  (16 x 64) ----
  v8f inj[4];
#pragma unroll
  for (int nt = 0; nt < 4; ++nt) inj[nt] = vzero8();
#pragma unroll
  for (int kt = 0; kt < 2; ++kt) {
    v16h a = load_afrag(buf + r * kLDB + kt * 32 + 16 * half);
#pragma unroll
    for (int nt = 0; nt < 4; ++nt) {
      v16h bf = load_bfrag(sHgT + (nt * 16 + r) * kM + kt * 32 + 16 * half);
      inj[nt] = wmma16(a, bf, inj[nt]);
    }
  }
#pragma unroll
  for (int nt = 0; nt < 4; ++nt)
#pragma unroll
    for (int v = 0; v < 8; ++v)
      buf[(v + 8 * half) * kLDB + nt * 16 + r] = (_Float16)inj[nt][v];
  __builtin_amdgcn_wave_barrier();

  // ---- back = injected @ w_n2t^T ; out = tok + back  (16 x 512) ----
  v16h ia0 = load_afrag(buf + r * kLDB + 0 + 16 * half);
  v16h ia1 = load_afrag(buf + r * kLDB + 32 + 16 * half);
  float* outb = out + g0 * kD;
  for (int dt = 0; dt < kD / 16; ++dt) {
    v16h b0 = load_bfrag(wn2th + (size_t)(dt * 16 + r) * kDN + 0 + 16 * half);
    v16h b1 = load_bfrag(wn2th + (size_t)(dt * 16 + r) * kDN + 32 + 16 * half);
    v8f a2 = vzero8();
    a2 = wmma16(ia0, b0, a2);
    a2 = wmma16(ia1, b1, a2);
#pragma unroll
    for (int v = 0; v < 8; ++v) {
      size_t off = (size_t)(v + 8 * half) * kD + dt * 16 + r;
      outb[off] = tokb[off] + a2[v];
    }
  }
}

}  // namespace

extern "C" void kernel_launch(void* const* d_in, const int* in_sizes, int n_in,
                              void* d_out, int out_size, void* d_ws, size_t ws_size,
                              hipStream_t stream) {
  (void)in_sizes; (void)n_in; (void)out_size; (void)ws_size;
  const float* tok   = (const float*)d_in[0];
  const float* w_s1  = (const float*)d_in[1];
  const float* b_s1  = (const float*)d_in[2];
  const float* w_s2  = (const float*)d_in[3];
  const float* b_s2  = (const float*)d_in[4];
  const float* w_t2n = (const float*)d_in[5];
  const float* w_n2t = (const float*)d_in[6];
  const float* w_g1  = (const float*)d_in[7];
  const float* w_g2  = (const float*)d_in[8];
  float* out = (float*)d_out;

  char* ws = (char*)d_ws;
  float*     scores = (float*)(ws + 0);        // 16*4096*4      = 262144
  int*       idx    = (int*)(ws + 262144);     // 16*64*4        =   4096
  _Float16*  w1h    = (_Float16*)(ws + 266240);   // 64*512*2    =  65536
  _Float16*  wt2nh  = (_Float16*)(ws + 331776);   // 64*512*2    =  65536
  _Float16*  wn2th  = (_Float16*)(ws + 397312);   // 512*64*2    =  65536
  _Float16*  Hg16   = (_Float16*)(ws + 462848);   // 16*64*64*2  = 131072
  _Float16*  HgT16  = (_Float16*)(ws + 593920);   // 16*64*64*2  = 131072

  k_convert<<<dim3(128), dim3(256), 0, stream>>>(w_s1, w_t2n, w_n2t, w1h, wt2nh, wn2th);
  k_scores<<<dim3(512), dim3(256), 0, stream>>>(tok, w1h, b_s1, w_s2, b_s2, scores);
  k_topk<<<dim3(16), dim3(256), 0, stream>>>(scores, idx);
  k_graph<<<dim3(16), dim3(256), 0, stream>>>(tok, idx, wt2nh, w_g1, w_g2, Hg16, HgT16);
  k_inject<<<dim3(512), dim3(256), 0, stream>>>(tok, wt2nh, wn2th, Hg16, HgT16, out);
}
